// gnnet_19533511262569
// MI455X (gfx1250) — matrix operand order repaired and checked by reading, not compiled, not run on hardware
//
#include <hip/hip_runtime.h>
#include <hip/hip_bf16.h>
#include <math.h>

#define N_NODES 100000
#define N_EDGES 200000
#define N_BATCH 2048
#define DIM     256
#define NLAYER  5

typedef __bf16 v16bf __attribute__((ext_vector_type(16)));
typedef __bf16 v8bf  __attribute__((ext_vector_type(8)));
typedef float  v8f   __attribute__((ext_vector_type(8)));

__device__ __forceinline__ __bf16 to_bf16(float f) {
    unsigned u = __builtin_bit_cast(unsigned, f);
    unsigned r = (u + 0x7FFFu + ((u >> 16) & 1u)) >> 16;
    return __builtin_bit_cast(__bf16, (unsigned short)r);
}

// ---------------------------------------------------------------- utilities
__global__ void zero_kernel(float* __restrict__ p, int n) {
    int i = blockIdx.x * blockDim.x + threadIdx.x;
    if (i < n) p[i] = 0.f;
}

// convert row-major [K][Nn] f32 weights into [Nn][K] bf16 (K contiguous)
__global__ void conv_w_kernel(const float* __restrict__ in, __bf16* __restrict__ out,
                              int K, int Nn) {
    int idx = blockIdx.x * blockDim.x + threadIdx.x;
    if (idx >= K * Nn) return;
    int n = idx / K, k = idx - n * K;
    out[(size_t)n * K + k] = to_bf16(in[(size_t)k * Nn + n]);
}

// ---------------------------------------------------------------- embedding
__global__ void node_init_kernel(const int* __restrict__ x,
                                 const float* __restrict__ emb1,
                                 const float* __restrict__ emb2,
                                 float* __restrict__ h) {
    int idx = blockIdx.x * blockDim.x + threadIdx.x;
    if (idx >= N_NODES * 64) return;
    int i = idx >> 6, c = idx & 63;
    int a = x[i * 2 + 0], b = x[i * 2 + 1];
    float4 va = ((const float4*)(emb1 + (size_t)a * DIM))[c];
    float4 vb = ((const float4*)(emb2 + (size_t)b * DIM))[c];
    float4 o; o.x = va.x + vb.x; o.y = va.y + vb.y; o.z = va.z + vb.z; o.w = va.w + vb.w;
    ((float4*)(h + (size_t)i * DIM))[c] = o;
}

// agg = h + self_e
__global__ void agg_init_kernel(const float* __restrict__ h,
                                const float* __restrict__ s1,  // edge_emb1[l][4]
                                const float* __restrict__ s2,  // edge_emb2[l][0]
                                float* __restrict__ agg) {
    int idx = blockIdx.x * blockDim.x + threadIdx.x;
    if (idx >= N_NODES * 64) return;
    int i = idx >> 6, c = idx & 63;
    float4 hv = ((const float4*)(h + (size_t)i * DIM))[c];
    float4 e1 = ((const float4*)s1)[c];
    float4 e2 = ((const float4*)s2)[c];
    float4 o; o.x = hv.x + e1.x + e2.x; o.y = hv.y + e1.y + e2.y;
    o.z = hv.z + e1.z + e2.z; o.w = hv.w + e1.w + e2.w;
    ((float4*)(agg + (size_t)i * DIM))[c] = o;
}

// agg[dst] += h[src] + edge_emb1[l][bt] + edge_emb2[l][bd]  (global f32 atomics)
__global__ void edge_scatter_kernel(const int* __restrict__ ei,
                                    const int* __restrict__ ea,
                                    const float* __restrict__ h,
                                    const float* __restrict__ e1,  // edge_emb1 + l*5*DIM
                                    const float* __restrict__ e2,  // edge_emb2 + l*3*DIM
                                    float* __restrict__ agg) {
    int idx = blockIdx.x * blockDim.x + threadIdx.x;
    if (idx >= N_EDGES * 64) return;
    int e = idx >> 6, c = idx & 63;
    int src = ei[e], dst = ei[N_EDGES + e];
    int bt = ea[e * 2 + 0], bd = ea[e * 2 + 1];
    float4 hv = ((const float4*)(h  + (size_t)src * DIM))[c];
    float4 v1 = ((const float4*)(e1 + (size_t)bt  * DIM))[c];
    float4 v2 = ((const float4*)(e2 + (size_t)bd  * DIM))[c];
    float* base = agg + (size_t)dst * DIM + c * 4;
    atomicAdd(base + 0, hv.x + v1.x + v2.x);
    atomicAdd(base + 1, hv.y + v1.y + v2.y);
    atomicAdd(base + 2, hv.z + v1.z + v2.z);
    atomicAdd(base + 3, hv.w + v1.w + v2.w);
}

// ---------------------------------------------------------------- fused MLP
// z = relu(agg@W1 + b1)@W2 + b2, intermediate (64x512) lives in LDS as bf16.
// Also accumulates per-column sum / sum-of-squares for batchnorm.
__launch_bounds__(256)
__global__ void fused_mlp_kernel(const float* __restrict__ agg,
                                 const __bf16* __restrict__ w1,  // [512][256] bf16
                                 const float* __restrict__ b1,   // [512]
                                 const __bf16* __restrict__ w2,  // [256][512] bf16
                                 const float* __restrict__ b2,   // [256]
                                 float* __restrict__ z,
                                 float* __restrict__ gsum,
                                 float* __restrict__ gsumq,
                                 int n_rows) {
    __shared__ __bf16 As[64][264];    // 64x256 A tile (+pad), 33 KB
    __shared__ __bf16 Z1s[64][520];   // 64x512 intermediate (+pad), 66 KB
    __shared__ float  s_sum[DIM];
    __shared__ float  s_sumq[DIM];

    const int t    = threadIdx.x;
    const int lane = t & 31;
    const int wv   = t >> 5;          // 8 waves
    const int row0 = blockIdx.x * 64;

    s_sum[t & 255] = 0.f; s_sumq[t & 255] = 0.f;

    // Phase A: stage A tile into LDS as bf16
    for (int i = 0; i < 16; ++i) {
        int idx = t + i * 256;        // float4 units, 64x64
        int r = idx >> 6, c4 = idx & 63;
        int grow = row0 + r; if (grow >= n_rows) grow = n_rows - 1;
        float4 v = ((const float4*)(agg + (size_t)grow * DIM))[c4];
        As[r][c4 * 4 + 0] = to_bf16(v.x);
        As[r][c4 * 4 + 1] = to_bf16(v.y);
        As[r][c4 * 4 + 2] = to_bf16(v.z);
        As[r][c4 * 4 + 3] = to_bf16(v.w);
    }
    __syncthreads();

    const int rt   = wv >> 1;         // row-tile 0..3 (16 rows each)
    const int ch   = wv & 1;          // column half
    const int nl   = lane & 15;
    const int kh   = lane >> 4;       // K-half select per ISA 16-bit layouts
    const int arow = rt * 16 + nl;

    // GEMM1: Z1 = relu(A @ W1 + b1)  -> LDS (bf16)
    for (int ct8 = 0; ct8 < 16; ++ct8) {
        const int ng = (ch * 16 + ct8) * 16 + nl;          // output col 0..511
        const __bf16* wcol = w1 + (size_t)ng * 256;
        __builtin_prefetch(wcol, 0, 3);
        v8f acc = {};
        for (int kk = 0; kk < 8; ++kk) {
            const int kb = kk * 32;
            v8bf alo = *(const v8bf*)&As[arow][kb + kh * 8];
            v8bf ahi = *(const v8bf*)&As[arow][kb + 16 + kh * 8];
            v16bf a;
            #pragma unroll
            for (int j = 0; j < 8; ++j) { a[j] = alo[j]; a[8 + j] = ahi[j]; }
            const int ks = kb + kh * 8;
            v8bf blo = *(const v8bf*)(wcol + ks);
            v8bf bhi = *(const v8bf*)(wcol + ks + 16);
            v16bf b;
            #pragma unroll
            for (int j = 0; j < 8; ++j) { b[j] = blo[j]; b[8 + j] = bhi[j]; }
            acc = __builtin_amdgcn_wmma_f32_16x16x32_bf16(false, a, false, b,
                                                          (short)0, acc, false, false);
        }
        const float bias = b1[ng];
        #pragma unroll
        for (int r = 0; r < 8; ++r) {
            float v = acc[r] + bias;
            v = v > 0.f ? v : 0.f;
            Z1s[rt * 16 + kh * 8 + r][ng] = to_bf16(v);
        }
    }
    __syncthreads();

    // GEMM2: z = Z1 @ W2 + b2  (+ BN partial stats)
    for (int ct8 = 0; ct8 < 8; ++ct8) {
        const int ng = (ch * 8 + ct8) * 16 + nl;           // output col 0..255
        const __bf16* wcol = w2 + (size_t)ng * 512;
        __builtin_prefetch(wcol, 0, 3);
        v8f acc = {};
        for (int kk = 0; kk < 16; ++kk) {
            const int kb = kk * 32;
            v8bf alo = *(const v8bf*)&Z1s[arow][kb + kh * 8];
            v8bf ahi = *(const v8bf*)&Z1s[arow][kb + 16 + kh * 8];
            v16bf a;
            #pragma unroll
            for (int j = 0; j < 8; ++j) { a[j] = alo[j]; a[8 + j] = ahi[j]; }
            const int ks = kb + kh * 8;
            v8bf blo = *(const v8bf*)(wcol + ks);
            v8bf bhi = *(const v8bf*)(wcol + ks + 16);
            v16bf b;
            #pragma unroll
            for (int j = 0; j < 8; ++j) { b[j] = blo[j]; b[8 + j] = bhi[j]; }
            acc = __builtin_amdgcn_wmma_f32_16x16x32_bf16(false, a, false, b,
                                                          (short)0, acc, false, false);
        }
        const float bias = b2[ng];
        float ps = 0.f, pq = 0.f;
        #pragma unroll
        for (int r = 0; r < 8; ++r) {
            const int grow = row0 + rt * 16 + kh * 8 + r;
            float v = acc[r] + bias;
            if (grow < n_rows) {
                z[(size_t)grow * DIM + ng] = v;
                ps += v; pq += v * v;
            }
        }
        atomicAdd(&s_sum[ng], ps);     // ds_add_f32
        atomicAdd(&s_sumq[ng], pq);
    }
    __syncthreads();
    atomicAdd(&gsum [t & 255], s_sum [t & 255]);
    atomicAdd(&gsumq[t & 255], s_sumq[t & 255]);
}

// ---------------------------------------------------------------- batchnorm
__global__ void bn_finalize_kernel(const float* __restrict__ gsum,
                                   const float* __restrict__ gsumq,
                                   float* __restrict__ mu, float* __restrict__ inv) {
    int d = blockIdx.x * blockDim.x + threadIdx.x;
    if (d >= DIM) return;
    float m = gsum[d] * (1.f / (float)N_NODES);
    float v = gsumq[d] * (1.f / (float)N_NODES) - m * m;
    mu[d] = m;
    inv[d] = rsqrtf(v + 1e-5f);
}

__global__ void bn_relu_kernel(const float* __restrict__ z,
                               const float* __restrict__ mu,
                               const float* __restrict__ inv,
                               const float* __restrict__ g,
                               const float* __restrict__ bb,
                               float* __restrict__ h) {
    int idx = blockIdx.x * blockDim.x + threadIdx.x;
    if (idx >= N_NODES * 64) return;
    int i = idx >> 6, c = idx & 63, d = c * 4;
    float4 v = ((const float4*)(z + (size_t)i * DIM))[c];
    float4 o;
    o.x = fmaxf((v.x - mu[d + 0]) * inv[d + 0] * g[d + 0] + bb[d + 0], 0.f);
    o.y = fmaxf((v.y - mu[d + 1]) * inv[d + 1] * g[d + 1] + bb[d + 1], 0.f);
    o.z = fmaxf((v.z - mu[d + 2]) * inv[d + 2] * g[d + 2] + bb[d + 2], 0.f);
    o.w = fmaxf((v.w - mu[d + 3]) * inv[d + 3] * g[d + 3] + bb[d + 3], 0.f);
    ((float4*)(h + (size_t)i * DIM))[c] = o;
}

// ---------------------------------------------------------------- pooling
__global__ void pool_count_kernel(const int* __restrict__ batch, float* __restrict__ cnt) {
    int i = blockIdx.x * blockDim.x + threadIdx.x;
    if (i < N_NODES) atomicAdd(&cnt[batch[i]], 1.f);
}

__global__ void pool_sum_kernel(const int* __restrict__ batch,
                                const float* __restrict__ h,
                                float* __restrict__ pooled) {
    int idx = blockIdx.x * blockDim.x + threadIdx.x;
    if (idx >= N_NODES * 64) return;
    int i = idx >> 6, c = idx & 63;
    int b = batch[i];
    float4 v = ((const float4*)(h + (size_t)i * DIM))[c];
    float* base = pooled + (size_t)b * DIM + c * 4;
    atomicAdd(base + 0, v.x);
    atomicAdd(base + 1, v.y);
    atomicAdd(base + 2, v.z);
    atomicAdd(base + 3, v.w);
}

// ---------------------------------------------------------------- head MLP
__global__ void head_kernel(const float* __restrict__ pooled,
                            const float* __restrict__ cnt,
                            const float* __restrict__ Wo1, const float* __restrict__ bo1,
                            const float* __restrict__ Wo2, const float* __restrict__ bo2,
                            float* __restrict__ out) {
    __shared__ float pr[DIM];
    __shared__ float hid[128];
    const int b = blockIdx.x;
    const int t = threadIdx.x;   // 128 threads
    const float inv = 1.f / fmaxf(cnt[b], 1.f);
    pr[t]       = pooled[(size_t)b * DIM + t]       * inv;
    pr[t + 128] = pooled[(size_t)b * DIM + t + 128] * inv;
    __syncthreads();
    float acc = bo1[t];
    for (int k = 0; k < DIM; ++k) acc += pr[k] * Wo1[k * 128 + t];
    hid[t] = (acc > 20.f) ? acc : log1pf(expf(acc));
    __syncthreads();
    if (t < 2) {
        float o = bo2[t];
        for (int k = 0; k < 128; ++k) o += hid[k] * Wo2[k * 2 + t];
        out[b * 2 + t] = o;
    }
}

// ---------------------------------------------------------------- launch
extern "C" void kernel_launch(void* const* d_in, const int* in_sizes, int n_in,
                              void* d_out, int out_size, void* d_ws, size_t ws_size,
                              hipStream_t stream) {
    (void)in_sizes; (void)n_in; (void)out_size; (void)ws_size;
    const int*   x      = (const int*)d_in[0];
    const int*   ei     = (const int*)d_in[1];
    const int*   ea     = (const int*)d_in[2];
    const int*   batch  = (const int*)d_in[3];
    const float* x_emb1 = (const float*)d_in[4];
    const float* x_emb2 = (const float*)d_in[5];
    const float* e_emb1 = (const float*)d_in[6];   // [5][5][256]
    const float* e_emb2 = (const float*)d_in[7];   // [5][3][256]
    const float* W1     = (const float*)d_in[8];   // [5][256][512]
    const float* b1     = (const float*)d_in[9];   // [5][512]
    const float* W2     = (const float*)d_in[10];  // [5][512][256]
    const float* b2     = (const float*)d_in[11];  // [5][256]
    const float* bn_g   = (const float*)d_in[12];
    const float* bn_b   = (const float*)d_in[13];
    const float* Wo1    = (const float*)d_in[14];
    const float* bo1    = (const float*)d_in[15];
    const float* Wo2    = (const float*)d_in[16];
    const float* bo2    = (const float*)d_in[17];
    float* out = (float*)d_out;

    char* ws = (char*)d_ws;
    const size_t OFF_H      = 0;
    const size_t OFF_Z      = 102400000;               // N*256*4
    const size_t OFF_W1     = 204800000;
    const size_t OFF_W2     = 206110720;               // +5*512*256*2
    const size_t OFF_STATS  = 207421440;               // +5*256*512*2
    const size_t OFF_POOLED = 207425536;               // +4096
    const size_t OFF_CNT    = 209522688;               // +B*256*4

    float*  h      = (float*)(ws + OFF_H);
    float*  zbuf   = (float*)(ws + OFF_Z);             // serves as agg AND z (row-disjoint)
    __bf16* w1bf   = (__bf16*)(ws + OFF_W1);
    __bf16* w2bf   = (__bf16*)(ws + OFF_W2);
    float*  stats  = (float*)(ws + OFF_STATS);
    float*  gsum   = stats;
    float*  gsumq  = stats + DIM;
    float*  mu     = stats + 2 * DIM;
    float*  inv    = stats + 3 * DIM;
    float*  pooled = (float*)(ws + OFF_POOLED);
    float*  counts = (float*)(ws + OFF_CNT);

    const int nd4   = N_NODES * 64;                    // float4 threads over N x D
    const int nGrid = (nd4 + 255) / 256;
    const int eGrid = (N_EDGES * 64 + 255) / 256;
    const int mGrid = (N_NODES + 63) / 64;

    // weight conversion to transposed bf16 (L2-resident)
    for (int l = 0; l < NLAYER; ++l) {
        conv_w_kernel<<<512, 256, 0, stream>>>(W1 + (size_t)l * 256 * 512,
                                               w1bf + (size_t)l * 512 * 256, 256, 512);
        conv_w_kernel<<<512, 256, 0, stream>>>(W2 + (size_t)l * 512 * 256,
                                               w2bf + (size_t)l * 256 * 512, 512, 256);
    }

    node_init_kernel<<<nGrid, 256, 0, stream>>>(x, x_emb1, x_emb2, h);

    for (int l = 0; l < NLAYER; ++l) {
        agg_init_kernel<<<nGrid, 256, 0, stream>>>(
            h, e_emb1 + ((size_t)l * 5 + 4) * DIM, e_emb2 + (size_t)l * 3 * DIM, zbuf);
        edge_scatter_kernel<<<eGrid, 256, 0, stream>>>(
            ei, ea, h, e_emb1 + (size_t)l * 5 * DIM, e_emb2 + (size_t)l * 3 * DIM, zbuf);
        zero_kernel<<<2, 256, 0, stream>>>(stats, 2 * DIM);
        fused_mlp_kernel<<<mGrid, 256, 0, stream>>>(
            zbuf, w1bf + (size_t)l * 512 * 256, b1 + (size_t)l * 512,
            w2bf + (size_t)l * 256 * 512, b2 + (size_t)l * 256,
            zbuf, gsum, gsumq, N_NODES);
        bn_finalize_kernel<<<1, 256, 0, stream>>>(gsum, gsumq, mu, inv);
        bn_relu_kernel<<<nGrid, 256, 0, stream>>>(
            zbuf, mu, inv, bn_g + (size_t)l * DIM, bn_b + (size_t)l * DIM, h);
    }

    zero_kernel<<<(N_BATCH * DIM + 255) / 256, 256, 0, stream>>>(pooled, N_BATCH * DIM);
    zero_kernel<<<(N_BATCH + 255) / 256, 256, 0, stream>>>(counts, N_BATCH);
    pool_count_kernel<<<(N_NODES + 255) / 256, 256, 0, stream>>>(batch, counts);
    pool_sum_kernel<<<nGrid, 256, 0, stream>>>(batch, h, pooled);
    head_kernel<<<N_BATCH, 128, 0, stream>>>(pooled, counts, Wo1, bo1, Wo2, bo2, out);
}
